// FET_70652212019634
// MI455X (gfx1250) — matrix operand order repaired and checked by reading, not compiled, not run on hardware
//
#include <hip/hip_runtime.h>
#include <math.h>

// ---------------- types ----------------
typedef __bf16 bf16;
typedef bf16  bf16x4  __attribute__((ext_vector_type(4)));
typedef bf16  bf16x8  __attribute__((ext_vector_type(8)));
typedef bf16  bf16x16 __attribute__((ext_vector_type(16)));
typedef float v8f     __attribute__((ext_vector_type(8)));

#define N_IMG   4
#define C_DIM   256
#define F_DIM   512
#define H_HEADS 8
#define D_HEAD  32
#define HW_TOK  16384
#define T_TOK   65536   // N_IMG * HW_TOK
#define L_LAYERS 4

// ---------------- small helpers ----------------
__global__ void cvt_f32_to_bf16(const float* __restrict__ src, bf16* __restrict__ dst, int n) {
    int i = blockIdx.x * 256 + threadIdx.x;
    if (i < n) dst[i] = (bf16)src[i];
}

__global__ void zero_f32(float* __restrict__ p, int n) {
    int i = blockIdx.x * 256 + threadIdx.x;
    if (i < n) p[i] = 0.0f;
}

// ref [N, C, HW] -> xf32/xb [T, C]   (T = n*HW + p)
__global__ void embed_kernel(const float* __restrict__ ref,
                             float* __restrict__ xf, bf16* __restrict__ xb) {
    size_t id = (size_t)blockIdx.x * 256 + threadIdx.x;   // over N*C*HW
    int    p  = (int)(id & (HW_TOK - 1));
    size_t q  = id >> 14;
    int    c  = (int)(q & (C_DIM - 1));
    int    n  = (int)(q >> 8);
    float  v  = ref[id];
    size_t t  = (size_t)n * HW_TOK + p;
    xf[t * C_DIM + c] = v;
    xb[t * C_DIM + c] = (bf16)v;
}

// xf32 [T, C] -> out [N, C, HW]  (one layer slice)
__global__ void store_out_kernel(const float* __restrict__ xf, float* __restrict__ out) {
    size_t id = (size_t)blockIdx.x * 256 + threadIdx.x;   // over N*C*HW
    int    p  = (int)(id & (HW_TOK - 1));
    size_t q  = id >> 14;
    int    c  = (int)(q & (C_DIM - 1));
    int    n  = (int)(q >> 8);
    out[id] = xf[((size_t)n * HW_TOK + p) * C_DIM + c];
}

// ---------------- WMMA GEMM, 64x64 tile per 4-wave block -----------------------
// D = A[T,K] x W[Nout,K]^T + bias, epilogue act: 0=none, 1=elu+1, 2=relu.
// Weight panel W[n0..n0+63, 0..kdim) staged once into LDS, shared by 4 waves;
// each wave = 16 rows x 64 cols = 4 accumulators, 4 WMMAs per 32-wide K-step.
__global__ void __launch_bounds__(128)
gemm64_bf16(const bf16* __restrict__ A, int lda,
            const bf16* __restrict__ W, int ldw,
            const float* __restrict__ bias,
            float* __restrict__ outF, bf16* __restrict__ outB, int ldo,
            int kdim, int act) {
    __shared__ bf16 Bs[64 * F_DIM];          // 64 KB max (kdim <= 512)

    int tid  = threadIdx.x;                  // 0..127
    int wave = tid >> 5;
    int lane = tid & 31;
    int l15  = lane & 15;
    int half = lane >> 4;
    int t0 = blockIdx.x * 64 + wave * 16;
    int n0 = blockIdx.y * 64;

    // cooperative stage of the 64-row weight panel into LDS (16B chunks)
    int chunks = (64 * kdim) >> 3;           // # of bf16x8 chunks (2048 or 4096)
    int cpr    = kdim >> 3;                  // chunks per row
    for (int i = tid; i < chunks; i += 128) {
        int row  = i / cpr;
        int col8 = i - row * cpr;
        *(bf16x8*)&Bs[row * kdim + col8 * 8] =
            *(const bf16x8*)(W + (size_t)(n0 + row) * ldw + col8 * 8);
    }
    __syncthreads();

    float b0 = bias[n0      + l15];
    float b1 = bias[n0 + 16 + l15];
    float b2 = bias[n0 + 32 + l15];
    float b3 = bias[n0 + 48 + l15];
    v8f acc0 = {b0, b0, b0, b0, b0, b0, b0, b0};
    v8f acc1 = {b1, b1, b1, b1, b1, b1, b1, b1};
    v8f acc2 = {b2, b2, b2, b2, b2, b2, b2, b2};
    v8f acc3 = {b3, b3, b3, b3, b3, b3, b3, b3};

    const bf16* arow = A + (size_t)(t0 + l15) * lda + half * 16;
    int ksteps = kdim >> 5;
    for (int kb = 0; kb < ksteps; ++kb) {
        bf16x8 alo = *(const bf16x8*)(arow + kb * 32);
        bf16x8 ahi = *(const bf16x8*)(arow + kb * 32 + 8);
        bf16x16 af = __builtin_shufflevector(alo, ahi,
                        0,1,2,3,4,5,6,7,8,9,10,11,12,13,14,15);
        int koff = kb * 32 + half * 16;
        bf16x16 bf0 = *(const bf16x16*)&Bs[(l15     ) * kdim + koff];
        bf16x16 bf1 = *(const bf16x16*)&Bs[(l15 + 16) * kdim + koff];
        bf16x16 bf2 = *(const bf16x16*)&Bs[(l15 + 32) * kdim + koff];
        bf16x16 bf3 = *(const bf16x16*)&Bs[(l15 + 48) * kdim + koff];
        acc0 = __builtin_amdgcn_wmma_f32_16x16x32_bf16(false, af, false, bf0, (short)0, acc0, false, false);
        acc1 = __builtin_amdgcn_wmma_f32_16x16x32_bf16(false, af, false, bf1, (short)0, acc1, false, false);
        acc2 = __builtin_amdgcn_wmma_f32_16x16x32_bf16(false, af, false, bf2, (short)0, acc2, false, false);
        acc3 = __builtin_amdgcn_wmma_f32_16x16x32_bf16(false, af, false, bf3, (short)0, acc3, false, false);
    }

    int mbase = half * 8;
    v8f accs[4] = {acc0, acc1, acc2, acc3};
    #pragma unroll
    for (int nt = 0; nt < 4; ++nt) {
        int ccol = n0 + nt * 16 + l15;
        #pragma unroll
        for (int i = 0; i < 8; ++i) {
            float v = accs[nt][i];
            if (act == 1)      v = (v > 0.0f) ? (v + 1.0f) : __expf(v);
            else if (act == 2) v = (v > 0.0f) ? v : 0.0f;
            size_t idx = (size_t)(t0 + mbase + i) * ldo + ccol;
            if (outF) outF[idx] = v;
            if (outB) outB[idx] = (bf16)v;
        }
    }
}

// --------- KV = sum_s v[s,m] * k[s,d]  per (n,h), plus Ksum[d]  (fp32 atomics) ---
__global__ void kv_reduce_kernel(const bf16* __restrict__ kb, const bf16* __restrict__ vb,
                                 float* __restrict__ kv, float* __restrict__ ksum) {
    int tid   = threadIdx.x;        // 256
    int chunk = blockIdx.x;         // 8 chunks of 2048 tokens
    int nh    = blockIdx.y;         // 32 = N*H
    int n = nh >> 3, h = nh & 7;
    int d  = tid & 31;
    int m0 = (tid >> 5) * 4;        // 0,4,...,28 ; thread owns m0..m0+3 (contiguous)
    size_t s0 = (size_t)n * HW_TOK + (size_t)chunk * 2048;
    int hb = h * D_HEAD;

    float a0 = 0, a1 = 0, a2 = 0, a3 = 0, ks = 0;
    #pragma unroll 4
    for (int s = 0; s < 2048; ++s) {
        size_t row = (s0 + s) * C_DIM + hb;
        float kvv = (float)kb[row + d];
        if (m0 == 0) ks += kvv;
        bf16x4 vv = *(const bf16x4*)(vb + row + m0);   // one 8-byte load
        a0 += (float)vv[0] * kvv;
        a1 += (float)vv[1] * kvv;
        a2 += (float)vv[2] * kvv;
        a3 += (float)vv[3] * kvv;
    }
    float* kvp = kv + (size_t)nh * (D_HEAD * D_HEAD);   // layout [m][d]
    atomicAdd(&kvp[(m0    ) * 32 + d], a0);
    atomicAdd(&kvp[(m0 + 1) * 32 + d], a1);
    atomicAdd(&kvp[(m0 + 2) * 32 + d], a2);
    atomicAdd(&kvp[(m0 + 3) * 32 + d], a3);
    if (m0 == 0) atomicAdd(&ksum[nh * 32 + d], ks);
}

// --------- Z[t,h] = 1 / (sum_d Q[t,h,d]*Ksum[n,h,d] + eps) ---------------------
__global__ void z_kernel(const bf16* __restrict__ qb, const float* __restrict__ ksum,
                         float* __restrict__ z) {
    int id = blockIdx.x * 256 + threadIdx.x;     // T * H
    int t = id >> 3, h = id & 7;
    int n = t >> 14;
    const bf16*  qp = qb + (size_t)t * C_DIM + h * D_HEAD;
    const float* kp = ksum + ((n * H_HEADS + h) << 5);
    float s = 1e-6f;
    #pragma unroll
    for (int d = 0; d < D_HEAD; ++d) s += (float)qp[d] * kp[d];
    z[id] = 1.0f / s;
}

// --------- attn[t, h*32+m] = Z[t,h] * sum_d Q[t,h,d] * KV[n,h,m,d]  (1 WMMA) ----
__global__ void __launch_bounds__(32)
attn16_kernel(const bf16* __restrict__ Q, const bf16* __restrict__ KVb,
              const float* __restrict__ Z, bf16* __restrict__ out) {
    int lane = threadIdx.x;
    int t0 = blockIdx.x * 16;
    int h  = blockIdx.y >> 1;
    int mt = (blockIdx.y & 1) * 16;
    int n  = t0 >> 14;
    int l15 = lane & 15, half = lane >> 4;

    const bf16* arow = Q + (size_t)(t0 + l15) * C_DIM + h * D_HEAD + half * 16;
    bf16x8 alo = *(const bf16x8*)(arow);
    bf16x8 ahi = *(const bf16x8*)(arow + 8);
    bf16x16 af = __builtin_shufflevector(alo, ahi,
                    0,1,2,3,4,5,6,7,8,9,10,11,12,13,14,15);
    const bf16* brow = KVb + (((size_t)(n * H_HEADS + h) * D_HEAD) + mt + l15) * D_HEAD
                           + half * 16;
    bf16x16 bf = *(const bf16x16*)brow;

    v8f acc = {0, 0, 0, 0, 0, 0, 0, 0};
    acc = __builtin_amdgcn_wmma_f32_16x16x32_bf16(
              false, af, false, bf, (short)0, acc, false, false);

    int mbase = half * 8;
    int ccol  = h * D_HEAD + mt + l15;
    #pragma unroll
    for (int i = 0; i < 8; ++i) {
        int t = t0 + mbase + i;
        out[(size_t)t * C_DIM + ccol] = (bf16)(acc[i] * Z[t * H_HEADS + h]);
    }
}

// --------- x = LN(xin + hbuf) * g + b ; also bf16 mirror --------------------------
__global__ void resid_ln_kernel(const float* __restrict__ xin, const float* __restrict__ hbuf,
                                const float* __restrict__ g, const float* __restrict__ b,
                                float* __restrict__ xo, bf16* __restrict__ xbo) {
    __shared__ float s1[C_DIM];
    __shared__ float s2[C_DIM];
    int t = blockIdx.x, c = threadIdx.x;
    size_t idx = (size_t)t * C_DIM + c;
    float v = xin[idx] + hbuf[idx];
    s1[c] = v; s2[c] = v * v;
    __syncthreads();
    for (int st = C_DIM / 2; st > 0; st >>= 1) {
        if (c < st) { s1[c] += s1[c + st]; s2[c] += s2[c + st]; }
        __syncthreads();
    }
    float mean = s1[0] * (1.0f / C_DIM);
    float var  = s2[0] * (1.0f / C_DIM) - mean * mean;
    float y = (v - mean) * rsqrtf(var + 1e-5f) * g[c] + b[c];
    xo[idx]  = y;
    xbo[idx] = (bf16)y;
}

// ================================================================================
extern "C" void kernel_launch(void* const* d_in, const int* in_sizes, int n_in,
                              void* d_out, int out_size, void* d_ws, size_t ws_size,
                              hipStream_t stream) {
    (void)in_sizes; (void)n_in; (void)out_size; (void)ws_size;
    const float* ref = (const float*)d_in[0];
    const float* Wq = (const float*)d_in[1];  const float* bq = (const float*)d_in[2];
    const float* Wk = (const float*)d_in[3];  const float* bk = (const float*)d_in[4];
    const float* Wv = (const float*)d_in[5];  const float* bv = (const float*)d_in[6];
    const float* Wo = (const float*)d_in[7];  const float* bo = (const float*)d_in[8];
    const float* g1 = (const float*)d_in[9];  const float* be1 = (const float*)d_in[10];
    const float* W1 = (const float*)d_in[11]; const float* c1 = (const float*)d_in[12];
    const float* W2 = (const float*)d_in[13]; const float* c2 = (const float*)d_in[14];
    const float* g2 = (const float*)d_in[15]; const float* be2 = (const float*)d_in[16];
    float* out = (float*)d_out;

    // ---- workspace carve-up ----
    char* ws = (char*)d_ws;
    size_t off = 0;
    auto carve = [&](size_t bytes) -> char* {
        char* p = ws + off;
        off += (bytes + 255) & ~(size_t)255;
        return p;
    };
    float* xf    = (float*)carve((size_t)T_TOK * C_DIM * 4);
    float* h1    = (float*)carve((size_t)T_TOK * C_DIM * 4);
    bf16*  xb    = (bf16*) carve((size_t)T_TOK * C_DIM * 2);
    bf16*  qb    = (bf16*) carve((size_t)T_TOK * C_DIM * 2);
    bf16*  kbuf  = (bf16*) carve((size_t)T_TOK * C_DIM * 2);
    bf16*  vbuf  = (bf16*) carve((size_t)T_TOK * C_DIM * 2);   // later reused for attn out
    bf16*  midb  = (bf16*) carve((size_t)T_TOK * F_DIM * 2);
    float* zbuf  = (float*)carve((size_t)T_TOK * H_HEADS * 4);
    float* kvf   = (float*)carve((size_t)N_IMG * H_HEADS * D_HEAD * D_HEAD * 4);
    bf16*  kvb16 = (bf16*) carve((size_t)N_IMG * H_HEADS * D_HEAD * D_HEAD * 2);
    float* ksum  = (float*)carve((size_t)N_IMG * H_HEADS * D_HEAD * 4);
    bf16*  wqb   = (bf16*) carve((size_t)L_LAYERS * C_DIM * C_DIM * 2);
    bf16*  wkb   = (bf16*) carve((size_t)L_LAYERS * C_DIM * C_DIM * 2);
    bf16*  wvb   = (bf16*) carve((size_t)L_LAYERS * C_DIM * C_DIM * 2);
    bf16*  wob   = (bf16*) carve((size_t)L_LAYERS * C_DIM * C_DIM * 2);
    bf16*  w1b   = (bf16*) carve((size_t)L_LAYERS * F_DIM * C_DIM * 2);
    bf16*  w2b   = (bf16*) carve((size_t)L_LAYERS * C_DIM * F_DIM * 2);

    const int nCC = L_LAYERS * C_DIM * C_DIM;          // 262144
    const int nFC = L_LAYERS * F_DIM * C_DIM;          // 524288

    // ---- convert all weights once ----
    cvt_f32_to_bf16<<<(nCC + 255) / 256, 256, 0, stream>>>(Wq, wqb, nCC);
    cvt_f32_to_bf16<<<(nCC + 255) / 256, 256, 0, stream>>>(Wk, wkb, nCC);
    cvt_f32_to_bf16<<<(nCC + 255) / 256, 256, 0, stream>>>(Wv, wvb, nCC);
    cvt_f32_to_bf16<<<(nCC + 255) / 256, 256, 0, stream>>>(Wo, wob, nCC);
    cvt_f32_to_bf16<<<(nFC + 255) / 256, 256, 0, stream>>>(W1, w1b, nFC);
    cvt_f32_to_bf16<<<(nFC + 255) / 256, 256, 0, stream>>>(W2, w2b, nFC);

    // ---- patch embed (transpose NCHW -> tokens) ----
    embed_kernel<<<(T_TOK * C_DIM) / 256, 256, 0, stream>>>(ref, xf, xb);

    dim3 gC(T_TOK / 64, C_DIM / 64);    // 1024 x 4  (256 outputs)
    dim3 gF(T_TOK / 64, F_DIM / 64);    // 1024 x 8  (512 outputs)
    const int kvN = N_IMG * H_HEADS * D_HEAD * D_HEAD;       // 32768
    const int ksN = N_IMG * H_HEADS * D_HEAD;                // 1024

    for (int l = 0; l < L_LAYERS; ++l) {
        const size_t wCC = (size_t)l * C_DIM * C_DIM;
        const size_t wFC = (size_t)l * F_DIM * C_DIM;

        // Q/K/V projections (Q,K with elu+1 feature map fused)
        gemm64_bf16<<<gC, 128, 0, stream>>>(xb, C_DIM, wqb + wCC, C_DIM, bq + l * C_DIM,
                                            nullptr, qb,   C_DIM, C_DIM, 1);
        gemm64_bf16<<<gC, 128, 0, stream>>>(xb, C_DIM, wkb + wCC, C_DIM, bk + l * C_DIM,
                                            nullptr, kbuf, C_DIM, C_DIM, 1);
        gemm64_bf16<<<gC, 128, 0, stream>>>(xb, C_DIM, wvb + wCC, C_DIM, bv + l * C_DIM,
                                            nullptr, vbuf, C_DIM, C_DIM, 0);

        // KV + Ksum reduction (tiny einsums, fp32 atomics)
        zero_f32<<<(kvN + 255) / 256, 256, 0, stream>>>(kvf, kvN);
        zero_f32<<<(ksN + 255) / 256, 256, 0, stream>>>(ksum, ksN);
        kv_reduce_kernel<<<dim3(8, N_IMG * H_HEADS), 256, 0, stream>>>(kbuf, vbuf, kvf, ksum);
        cvt_f32_to_bf16<<<(kvN + 255) / 256, 256, 0, stream>>>(kvf, kvb16, kvN);

        // Z normalizer, then attn = Z * (Q @ KV)  -> overwrites vbuf
        z_kernel<<<(T_TOK * H_HEADS) / 256, 256, 0, stream>>>(qb, ksum, zbuf);
        attn16_kernel<<<dim3(T_TOK / 16, H_HEADS * 2), 32, 0, stream>>>(qb, kvb16, zbuf, vbuf);

        // output projection + residual + LN1
        gemm64_bf16<<<gC, 128, 0, stream>>>(vbuf, C_DIM, wob + wCC, C_DIM, bo + l * C_DIM,
                                            h1, nullptr, C_DIM, C_DIM, 0);
        resid_ln_kernel<<<T_TOK, C_DIM, 0, stream>>>(xf, h1, g1 + l * C_DIM, be1 + l * C_DIM,
                                                     xf, xb);

        // FFN: relu(x @ W1^T + c1) @ W2^T + c2, residual + LN2
        gemm64_bf16<<<gF, 128, 0, stream>>>(xb, C_DIM, w1b + wFC, C_DIM, c1 + l * F_DIM,
                                            nullptr, midb, F_DIM, C_DIM, 2);
        gemm64_bf16<<<gC, 128, 0, stream>>>(midb, F_DIM, w2b + wFC, F_DIM, c2 + l * C_DIM,
                                            h1, nullptr, C_DIM, F_DIM, 0);
        resid_ln_kernel<<<T_TOK, C_DIM, 0, stream>>>(xf, h1, g2 + l * C_DIM, be2 + l * C_DIM,
                                                     xf, xb);

        // emit this layer's x as [N, C, H, W]
        store_out_kernel<<<(T_TOK * C_DIM) / 256, 256, 0, stream>>>(
            xf, out + (size_t)l * T_TOK * C_DIM);
    }
}
